// CorrelationAttentionLayer_45621142618792
// MI455X (gfx1250) — matrix operand order repaired and checked by reading, not compiled, not run on hardware
//
#include <hip/hip_runtime.h>

#define kB 8
#define kL 4096
#define kD 512
#define kTOPK 5

typedef __attribute__((ext_vector_type(2)))  float  v2f;
typedef __attribute__((ext_vector_type(8)))  float  v8f;
typedef __attribute__((ext_vector_type(16))) __bf16 v16bf;

__device__ __forceinline__ unsigned short f2bf(float f) {
  unsigned u = __builtin_bit_cast(unsigned, f);
  unsigned r = u + 0x7FFFu + ((u >> 16) & 1u);   // round-to-nearest-even
  return (unsigned short)(r >> 16);
}

// ---------------------------------------------------------------------------
// Zero-init kernel (mean_value is an atomic accumulation target)
// ---------------------------------------------------------------------------
__global__ void zero_f32_kernel(float* p, int n) {
  int i = blockIdx.x * 256 + threadIdx.x;
  if (i < n) p[i] = 0.0f;
}

// ---------------------------------------------------------------------------
// Projection GEMM: out[m,n] = sum_k X[m,k]*W[n,k] + bias[n]
// M = 32768, N = K = 512.  2x2 register blocking: one 32x32 output block per
// wave, f32 WMMA (16x16x4), 4 loads feed 4 WMMAs per K-step.
// OUT_BF16=1 -> store raw bf16 bits (for Q,K); OUT_BF16=0 -> f32 (for V).
// ---------------------------------------------------------------------------
template <int OUT_BF16>
__global__ void proj_gemm_kernel(const float* __restrict__ X,
                                 const float* __restrict__ W,
                                 const float* __restrict__ bias,
                                 void* __restrict__ outp) {
  const int wave = threadIdx.x >> 5;
  const int lane = threadIdx.x & 31;
  const int wtid = blockIdx.x * 8 + wave;       // 16384 blocks of 32x32
  const int nt = wtid & 15;                     // N/32 = 16
  const int mt = wtid >> 4;                     // M/32 = 1024
  const int m0 = mt * 32, n0 = nt * 32;
  const int hsel = lane >> 4;                   // K-half select
  const int lid  = lane & 15;                   // row (A) / col (B)

  const float* ar[2];
  const float* br[2];
#pragma unroll
  for (int p = 0; p < 2; ++p)
    ar[p] = X + (size_t)(m0 + p * 16 + lid) * kD + hsel * 2;
#pragma unroll
  for (int q = 0; q < 2; ++q)
    br[q] = W + (size_t)(n0 + q * 16 + lid) * kD + hsel * 2;

  v8f acc[2][2] = {};
#pragma unroll 4
  for (int kk = 0; kk < kD; kk += 4) {
    v2f a0 = *(const v2f*)(ar[0] + kk);
    v2f a1 = *(const v2f*)(ar[1] + kk);
    v2f b0 = *(const v2f*)(br[0] + kk);
    v2f b1 = *(const v2f*)(br[1] + kk);
    acc[0][0] = __builtin_amdgcn_wmma_f32_16x16x4_f32(false, a0, false, b0, (short)0, acc[0][0], false, false);
    acc[0][1] = __builtin_amdgcn_wmma_f32_16x16x4_f32(false, a0, false, b1, (short)0, acc[0][1], false, false);
    acc[1][0] = __builtin_amdgcn_wmma_f32_16x16x4_f32(false, a1, false, b0, (short)0, acc[1][0], false, false);
    acc[1][1] = __builtin_amdgcn_wmma_f32_16x16x4_f32(false, a1, false, b1, (short)0, acc[1][1], false, false);
  }

#pragma unroll
  for (int p = 0; p < 2; ++p) {
#pragma unroll
    for (int q = 0; q < 2; ++q) {
      const float bn = bias[n0 + q * 16 + lid];
      const size_t obase =
          (size_t)(m0 + p * 16 + hsel * 8) * kD + (n0 + q * 16 + lid);
      if (OUT_BF16) {
        unsigned short* o = (unsigned short*)outp;
#pragma unroll
        for (int r = 0; r < 8; ++r)
          o[obase + (size_t)r * kD] = f2bf(acc[p][q][r] + bn);
      } else {
        float* o = (float*)outp;
#pragma unroll
        for (int r = 0; r < 8; ++r)
          o[obase + (size_t)r * kD] = acc[p][q][r] + bn;
      }
    }
  }
}

// ---------------------------------------------------------------------------
// Correlation: mean_value[b,tau] = (1/D) * sum_t <Q[b,t,:], K[b,(t-tau)%L,:]>
// Tiled Q*K^T (bf16 WMMA 16x16x32) with 4x2 register blocking (64x32 output
// per wave; 6 operand loads feed 8 WMMAs per K-step).  Each 16x16 sub-tile is
// reduced over its 31 circular diagonals in wave-private LDS slots, then the
// partials are atomically accumulated into mean_value.
// ---------------------------------------------------------------------------
__global__ void corr_gemm_kernel(const unsigned short* __restrict__ Qb,
                                 const unsigned short* __restrict__ Kb,
                                 float* __restrict__ mean_value) {
  __shared__ float diag[8][8][32];              // [wave][subtile][slot]
  const int wave = threadIdx.x >> 5;
  const int lane = threadIdx.x & 31;
  const int wtid = blockIdx.x * 8 + wave;       // 65536 waves total
  const int b    = wtid >> 13;                  // 8192 blocks per batch
  const int rem  = wtid & 8191;
  const int ti   = rem >> 7;                    // 64 row-groups (64 rows)
  const int sj   = rem & 127;                   // 128 col-groups (32 cols)
  const int hsel = lane >> 4;
  const int lid  = lane & 15;

  const unsigned short* qp[4];
  const unsigned short* kp[2];
#pragma unroll
  for (int p = 0; p < 4; ++p)
    qp[p] = Qb + ((size_t)b * kL + ti * 64 + p * 16 + lid) * kD;
#pragma unroll
  for (int q = 0; q < 2; ++q)
    kp[q] = Kb + ((size_t)b * kL + sj * 32 + q * 16 + lid) * kD;

  v8f acc[4][2] = {};
  for (int kk = 0; kk < kD; kk += 32) {
    union U { uint4 u[2]; v16bf v; };
    U A[4], Bv[2];
#pragma unroll
    for (int p = 0; p < 4; ++p) {
      // A (16x32 bf16): j<8 -> k = kk+hsel*8+j ; j>=8 -> k = kk+16+hsel*8+(j-8)
      A[p].u[0] = *(const uint4*)(qp[p] + kk + hsel * 8);
      A[p].u[1] = *(const uint4*)(qp[p] + kk + 16 + hsel * 8);
    }
#pragma unroll
    for (int q = 0; q < 2; ++q) {
      // B (32x16 bf16): j<8 -> k = kk+hsel*8+j ; j>=8 -> k = kk+16+hsel*8+(j-8)
      Bv[q].u[0] = *(const uint4*)(kp[q] + kk + hsel * 8);
      Bv[q].u[1] = *(const uint4*)(kp[q] + kk + 16 + hsel * 8);
    }
#pragma unroll
    for (int p = 0; p < 4; ++p)
#pragma unroll
      for (int q = 0; q < 2; ++q)
        acc[p][q] = __builtin_amdgcn_wmma_f32_16x16x32_bf16(
            false, A[p].v, false, Bv[q].v, (short)0, acc[p][q], false, false);
  }

  // Zero wave-private diagonal slots (per-wave LDS pipe is in-order).
#pragma unroll
  for (int st = 0; st < 8; ++st) diag[wave][st][lane] = 0.0f;

  // Scatter accumulator entries onto their local (t-s) diagonals.
#pragma unroll
  for (int p = 0; p < 4; ++p) {
#pragma unroll
    for (int q = 0; q < 2; ++q) {
      const int st = p * 2 + q;
#pragma unroll
      for (int r = 0; r < 8; ++r) {
        const int lt = hsel * 8 + r;            // local t of acc element
        atomicAdd(&diag[wave][st][lt - lid + 15], acc[p][q][r]);
      }
    }
  }

  // Flush the 31 diagonal partials of each sub-tile to mean_value.
  if (lane < 31) {
    const float inv = 1.0f / (float)kD;
#pragma unroll
    for (int p = 0; p < 4; ++p) {
#pragma unroll
      for (int q = 0; q < 2; ++q) {
        const int st  = p * 2 + q;
        const int tau = (((ti * 4 + p) - (sj * 2 + q)) * 16 + lane - 15) & (kL - 1);
        atomicAdd(&mean_value[b * kL + tau], diag[wave][st][lane] * inv);
      }
    }
  }
}

// ---------------------------------------------------------------------------
// Batch-mean top-k + per-batch softmax weights. Single block, 256 threads.
// ---------------------------------------------------------------------------
__global__ void topk_softmax_kernel(const float* __restrict__ mean_value,
                                    int* __restrict__ idx_out,
                                    float* __restrict__ w_out) {
  __shared__ float bmean[kL];
  __shared__ int   sidx[kTOPK];
  __shared__ float redv[256];
  __shared__ int   redi[256];
  const int tid = threadIdx.x;

  for (int l = tid; l < kL; l += 256) {
    float s = 0.0f;
    for (int b = 0; b < kB; ++b) s += mean_value[b * kL + l];
    bmean[l] = s;
  }
  __syncthreads();

  for (int i = 0; i < kTOPK; ++i) {
    float best = -3.4e38f; int bi = 0;
    for (int l = tid; l < kL; l += 256) {
      const float v = bmean[l];
      if (v > best) { best = v; bi = l; }
    }
    redv[tid] = best; redi[tid] = bi;
    __syncthreads();
    for (int s = 128; s > 0; s >>= 1) {
      if (tid < s && redv[tid + s] > redv[tid]) {
        redv[tid] = redv[tid + s]; redi[tid] = redi[tid + s];
      }
      __syncthreads();
    }
    if (tid == 0) {
      sidx[i] = redi[0];
      idx_out[i] = redi[0];
      bmean[redi[0]] = -3.4e38f;
    }
    __syncthreads();
  }

  if (tid < kB) {
    float w[kTOPK]; float mx = -3.4e38f;
    for (int i = 0; i < kTOPK; ++i) {
      w[i] = mean_value[tid * kL + sidx[i]];
      mx = fmaxf(mx, w[i]);
    }
    float sum = 0.0f;
    for (int i = 0; i < kTOPK; ++i) { w[i] = __expf(w[i] - mx); sum += w[i]; }
    const float inv = 1.0f / sum;
    for (int i = 0; i < kTOPK; ++i) w_out[tid * 8 + i] = w[i] * inv;
  }
}

// ---------------------------------------------------------------------------
// Aggregation: out[b,l,:] = sum_i w[b,i] * Vp[b, (l+idx[i]) % L, :]
// One row per block; 128 threads x float4.
// ---------------------------------------------------------------------------
__global__ void aggregate_kernel(const float* __restrict__ Vp,
                                 const int* __restrict__ idx,
                                 const float* __restrict__ w,
                                 float* __restrict__ out) {
  const int row = blockIdx.x;                   // 0 .. B*L-1
  const int b = row >> 12;                      // L = 4096
  const int l = row & (kL - 1);
  const int tid = threadIdx.x;

  float4 acc = make_float4(0.f, 0.f, 0.f, 0.f);
#pragma unroll
  for (int i = 0; i < kTOPK; ++i) {
    const int src = (l + idx[i]) & (kL - 1);
    const float4 v = ((const float4*)&Vp[((size_t)b * kL + src) * kD])[tid];
    const float wi = w[b * 8 + i];
    acc.x += wi * v.x; acc.y += wi * v.y; acc.z += wi * v.z; acc.w += wi * v.w;
  }
  ((float4*)&out[(size_t)row * kD])[tid] = acc;
}

// ---------------------------------------------------------------------------
extern "C" void kernel_launch(void* const* d_in, const int* in_sizes, int n_in,
                              void* d_out, int out_size, void* d_ws, size_t ws_size,
                              hipStream_t stream) {
  (void)in_sizes; (void)n_in; (void)out_size; (void)ws_size;
  const float* q  = (const float*)d_in[0];
  const float* k  = (const float*)d_in[1];
  const float* v  = (const float*)d_in[2];
  // d_in[3] = attention_mask (all-ones; unused by the reference math)
  const float* Wq = (const float*)d_in[4];
  const float* bq = (const float*)d_in[5];
  const float* Wk = (const float*)d_in[6];
  const float* bk = (const float*)d_in[7];
  const float* Wv = (const float*)d_in[8];
  const float* bv = (const float*)d_in[9];
  float* out = (float*)d_out;

  char* ws = (char*)d_ws;
  const size_t nTok = (size_t)kB * kL;                  // 32768
  float*          Vp  = (float*)ws;                     // 64 MB
  unsigned short* Qbf = (unsigned short*)(ws + nTok * kD * 4);            // 32 MB
  unsigned short* Kbf = (unsigned short*)(ws + nTok * kD * 4 + nTok * kD * 2);
  float* mean_value = (float*)(ws + nTok * kD * 8);     // B*L f32
  int*   topk_idx   = (int*)(ws + nTok * kD * 8 + nTok * 4);
  float* tmp_corr   = (float*)(ws + nTok * kD * 8 + nTok * 4 + 64);

  // 1) zero the atomic accumulator
  zero_f32_kernel<<<(int)((nTok + 255) / 256), 256, 0, stream>>>(mean_value, (int)nTok);

  // 2) projections: (M/32)*(N/32)/8 waves-per-block = 2048 blocks
  proj_gemm_kernel<0><<<2048, 256, 0, stream>>>(v, Wv, bv, (void*)Vp);
  proj_gemm_kernel<1><<<2048, 256, 0, stream>>>(q, Wq, bq, (void*)Qbf);
  proj_gemm_kernel<1><<<2048, 256, 0, stream>>>(k, Wk, bk, (void*)Kbf);

  // 3) autocorrelation via tiled Q*K^T with diagonal reduction
  corr_gemm_kernel<<<8192, 256, 0, stream>>>(Qbf, Kbf, mean_value);

  // 4) top-k over batch-mean + per-batch softmax weights
  topk_softmax_kernel<<<1, 256, 0, stream>>>(mean_value, topk_idx, tmp_corr);

  // 5) weighted circular-shift aggregation of V
  aggregate_kernel<<<(int)nTok, 128, 0, stream>>>(Vp, topk_idx, tmp_corr, out);
}
